// SelfAttention_1889785610832
// MI455X (gfx1250) — matrix operand order repaired and checked by reading, not compiled, not run on hardware
//
#include <hip/hip_runtime.h>
#include <hip/hip_bf16.h>
#include <stdint.h>

// ---------------- problem constants ----------------
constexpr int BB   = 2;
constexpr int NSEQ = 2048;
constexpr int DM   = 1024;
constexpr int HH   = 16;
constexpr int DK   = 64;
constexpr int QT   = NSEQ / 16;   // 128 q-tiles per (b,h)

typedef _Float16 half_t;
typedef __attribute__((ext_vector_type(16))) _Float16 v16h;
typedef __attribute__((ext_vector_type(8)))  float    v8f;

#define NEG_INF (-__builtin_inff())

union V16 { v16h v; uint32_t u[8]; };

// ---- WMMA operand loaders (contiguous k order, same for A and B, wave32) ----
// A 16x32 f16: lane m(0-15)=row m; VGPR v holds K pair:
//   k = (lane>=16 ? 16 : 0) + v*2
__device__ inline v16h loadA16x32(const half_t* __restrict__ X, int ldx, int lane) {
  int row  = lane & 15;
  int koff = (lane >> 4) << 4;           // 0 or 16
  V16 r;
  const half_t* base = X + (size_t)row * ldx;
#pragma unroll
  for (int v = 0; v < 8; ++v) {
    int k = koff + (v << 1);
    r.u[v] = *(const uint32_t*)(base + k);
  }
  return r.v;
}

// B 32x16 f16 where logical B[k][n] = W[n][k] (W row-major over k, pre-offset
// to (n0,k0)). Lane n(0-15)=col n; VGPR v holds pair k = (lane>=16?16:0)+2v.
// The 8 dwords are 32 contiguous bytes -> merges into (ds_/global_)load_b128 x2.
__device__ inline v16h loadBT(const half_t* W, int ldw, int lane) {
  int col = lane & 15;
  int kb  = (lane >> 4) << 4;            // 0 or 16
  V16 r;
  const half_t* base = W + (size_t)col * ldw + kb;
#pragma unroll
  for (int v = 0; v < 8; ++v)
    r.u[v] = *(const uint32_t*)(base + 2 * v);
  return r.v;
}

__device__ inline v8f wmma_f16(v16h a, v16h b, v8f c) {
  return __builtin_amdgcn_wmma_f32_16x16x32_f16(false, a, false, b, (short)0, c,
                                                false, false);
}

// ---- CDNA5 async global->LDS copy (ASYNCcnt path, ISA §10.7/§15.18) ----
// LDS byte address = low 32 bits of the generic pointer (ISA §10.2 aperture).
__device__ inline void async_g2l_b128(uint32_t lds_off, const char* g) {
  asm volatile("global_load_async_to_lds_b128 %0, %1, off"
               :: "v"(lds_off), "v"(g) : "memory");
}
__device__ inline void wait_asynccnt0() {
  asm volatile("s_wait_asynccnt 0x0" ::: "memory");
}
__device__ inline uint32_t lds_addr32(const void* p) {
  return (uint32_t)(uintptr_t)p;
}

// ---------------- f32 -> f16 convert ----------------
__global__ void f32_to_f16(const float* __restrict__ src, half_t* __restrict__ dst, int n) {
  int i = blockIdx.x * blockDim.x + threadIdx.x;
  if (i < n) dst[i] = (half_t)src[i];
}

// ---------------- GEMM: D = X @ W^T + bias ----------------
// X: M x K f16 row-major, W: N x K f16 row-major.
// Block = 4 waves sharing one 64-col W panel; the panel (64x32 f16, 4 KB) is
// double-buffered in LDS: async fill of tile k+32 overlaps WMMA on tile k.
// mode 0: f16 out, per-head (b,h,seq,dk)   (Q, K)
// mode 1: f16 out, per-head transposed (b,h,dk,seq)   (V)
// mode 2: f32 out, row-major M x N   (final projection)
#define GEMM_WAVES 4
__global__ __launch_bounds__(32 * GEMM_WAVES)
void gemm_xwt(const half_t* __restrict__ X, const half_t* __restrict__ Wm,
              const float* __restrict__ bias, void* __restrict__ outp,
              int M, int Nn, int K, int mode) {
  __shared__ half_t ldsW[2][64 * 32];    // [buf][n_row 0..63][k 0..31]

  int lane  = threadIdx.x & 31;
  int wave  = threadIdx.x >> 5;
  int m0    = (blockIdx.y * GEMM_WAVES + wave) * 16;
  int n0    = blockIdx.x * 64;
  uint32_t ldsWa[2] = { lds_addr32(&ldsW[0][0]), lds_addr32(&ldsW[1][0]) };

  // async fill of W panel [n0..n0+63] x [k..k+31]: 256 x b128, 2 per thread
  auto fillW = [&](int buf, int k) {
#pragma unroll
    for (int i = 0; i < 2; ++i) {
      int t   = threadIdx.x + i * 128;
      int row = t >> 2, seg = (t & 3) * 16;
      async_g2l_b128(ldsWa[buf] + row * 64 + seg,
                     (const char*)(Wm + (size_t)(n0 + row) * K + k) + seg);
    }
  };

  fillW(0, 0);
  v8f acc[4] = {};
  int it = 0;
  for (int k = 0; k < K; k += 32, ++it) {
    int cur = it & 1;
    wait_asynccnt0();                    // my fills for buf[cur] complete
    __syncthreads();                     // all waves: buf[cur] ready, buf[!cur] free
    if (k + 32 < K) fillW(cur ^ 1, k + 32);   // stream next tile behind WMMAs
    v16h a = loadA16x32(X + (size_t)m0 * K + k, K, lane);
#pragma unroll
    for (int j = 0; j < 4; ++j) {
      v16h b = loadBT(&ldsW[cur][(j * 16) * 32], 32, lane);
      acc[j] = wmma_f16(a, b, acc[j]);
    }
  }

  int colL = lane & 15;
  int coff = (lane >> 4) << 3;           // C layout: row = coff + v, col = colL
  if (mode == 2) {
    float* out = (float*)outp;
#pragma unroll
    for (int j = 0; j < 4; ++j) {
      int n = n0 + j * 16 + colL;
      float bv = bias[n];
#pragma unroll
      for (int v = 0; v < 8; ++v)
        out[(size_t)(m0 + coff + v) * Nn + n] = acc[j][v] + bv;
    }
  } else {
    half_t* out = (half_t*)outp;
#pragma unroll
    for (int j = 0; j < 4; ++j) {
      int n  = n0 + j * 16 + colL;
      int h  = n >> 6, dk = n & 63;
      float bv = bias[n];
#pragma unroll
      for (int v = 0; v < 8; ++v) {
        int m = m0 + coff + v;
        int b = m >> 11, s = m & (NSEQ - 1);
        size_t idx = (mode == 0)
          ? ((size_t)((b * HH + h) * NSEQ + s)) * DK + dk
          : ((size_t)((b * HH + h) * DK  + dk)) * NSEQ + s;
        out[idx] = (half_t)(acc[j][v] + bv);
      }
    }
  }
}

// ---------------- fused attention (flash-style, online softmax) ----------------
// One wave owns a 16-row q-tile; the 4 waves of a block share one (b,h), so the
// K tile (32x64) and V^T tile (64x32) are double-buffered in LDS via async
// loads: tile kb+32 streams in behind the WMMAs on tile kb. attw/mask (the
// 670 MB roofline stream) stays direct with a prefetch one tile ahead.
#define ATT_WAVES 4
__global__ __launch_bounds__(32 * ATT_WAVES)
void attn_kernel(const half_t* __restrict__ Q, const half_t* __restrict__ Km,
                 const half_t* __restrict__ Vt, const float* __restrict__ attw,
                 const unsigned char* __restrict__ mask, half_t* __restrict__ ctx) {
  __shared__ half_t ldsK[2][32 * 64];              // [buf][key 0..31][dk 0..63]
  __shared__ half_t ldsV[2][64 * 32];              // [buf][dk 0..63][key 0..31]
  __shared__ half_t plds[ATT_WAVES][16 * 32];

  int lane = threadIdx.x & 31;
  int wave = threadIdx.x >> 5;
  int wg   = blockIdx.x * ATT_WAVES + wave;
  int qtile = wg & (QT - 1);
  int bh    = wg >> 7;                   // QT == 128; blocks never straddle bh
  int q0    = qtile * 16;

  const half_t* Qbh = Q  + (size_t)bh * NSEQ * DK;
  const half_t* Kbh = Km + (size_t)bh * NSEQ * DK;
  const half_t* Vbh = Vt + (size_t)bh * DK * NSEQ;
  const float*  wbh = attw + (size_t)bh * NSEQ * NSEQ;
  const unsigned char* mbh = mask + (size_t)bh * NSEQ * NSEQ;

  uint32_t ldsKa[2] = { lds_addr32(&ldsK[0][0]), lds_addr32(&ldsK[1][0]) };
  uint32_t ldsVa[2] = { lds_addr32(&ldsV[0][0]), lds_addr32(&ldsV[1][0]) };

  // async fill of K (4 KB contiguous) + V^T (64 strided rows) tiles at key kb
  auto fillKV = [&](int buf, int kb) {
#pragma unroll
    for (int i = 0; i < 2; ++i) {
      int off = (threadIdx.x + i * 128) * 16;
      async_g2l_b128(ldsKa[buf] + off, (const char*)(Kbh + (size_t)kb * DK) + off);
    }
#pragma unroll
    for (int i = 0; i < 2; ++i) {
      int t   = threadIdx.x + i * 128;
      int row = t >> 2, seg = (t & 3) * 16;
      async_g2l_b128(ldsVa[buf] + row * 64 + seg,
                     (const char*)(Vbh + (size_t)row * NSEQ + kb) + seg);
    }
  };

  int colL = lane & 15;
  int coff = (lane >> 4) << 3;

  // Q tile 16x64 stays in registers (two 16x32 A operands)
  v16h qa0 = loadA16x32(Qbh + (size_t)q0 * DK,      DK, lane);
  v16h qa1 = loadA16x32(Qbh + (size_t)q0 * DK + 32, DK, lane);

  fillKV(0, 0);
  v8f acc[4] = {};
  float rm[8], rl[8];
#pragma unroll
  for (int v = 0; v < 8; ++v) { rm[v] = NEG_INF; rl[v] = 0.f; }
  const float scale = 0.125f;            // 1/sqrt(64)

  int it = 0;
  for (int kb = 0; kb < NSEQ; kb += 32, ++it) {
    int cur = it & 1;
    wait_asynccnt0();                    // my fills for buf[cur] complete
    __syncthreads();                     // all waves: buf[cur] ready, buf[!cur] free
    if (kb + 32 < NSEQ) {
      fillKV(cur ^ 1, kb + 32);          // stream next K/V tiles behind compute
      __builtin_prefetch(&wbh[(size_t)(q0 + coff) * NSEQ + kb + 32], 0, 0);
    }
    const half_t* lk = &ldsK[cur][0];
    const half_t* lv = &ldsV[cur][0];

    // ---- scores: two 16x16 tiles (keys kb..+15, kb+16..+31), from LDS ----
    v8f s0 = {}, s1 = {};
    {
      v16h b0 = loadBT(lk,                DK, lane);
      v16h b1 = loadBT(lk + 32,           DK, lane);
      s0 = wmma_f16(qa0, b0, s0); s0 = wmma_f16(qa1, b1, s0);
    }
    {
      v16h b0 = loadBT(lk + 16 * DK,      DK, lane);
      v16h b1 = loadBT(lk + 16 * DK + 32, DK, lane);
      s1 = wmma_f16(qa0, b0, s1); s1 = wmma_f16(qa1, b1, s1);
    }

    // ---- scale * attw, mask, online softmax (row reduce over 16 lanes) ----
    float p0[8], p1[8], al[8];
#pragma unroll
    for (int v = 0; v < 8; ++v) {
      size_t base = (size_t)(q0 + coff + v) * NSEQ + kb + colL;
      float t0 = s0[v] * scale * wbh[base];
      float t1 = s1[v] * scale * wbh[base + 16];
      t0 = mbh[base]      ? NEG_INF : t0;
      t1 = mbh[base + 16] ? NEG_INF : t1;
      float tmax = fmaxf(t0, t1);
#pragma unroll
      for (int s = 1; s < 16; s <<= 1)
        tmax = fmaxf(tmax, __shfl_xor(tmax, s, 32));
      float nm = fmaxf(rm[v], tmax);
      float alpha, e0, e1;
      if (nm == NEG_INF) { alpha = 1.f; e0 = 0.f; e1 = 0.f; }
      else {
        alpha = (rm[v] == NEG_INF) ? 0.f : __expf(rm[v] - nm);
        e0 = (t0 == NEG_INF) ? 0.f : __expf(t0 - nm);
        e1 = (t1 == NEG_INF) ? 0.f : __expf(t1 - nm);
      }
      float ps = e0 + e1;
#pragma unroll
      for (int s = 1; s < 16; s <<= 1)
        ps += __shfl_xor(ps, s, 32);
      rl[v] = rl[v] * alpha + ps;
      rm[v] = nm;
      al[v] = alpha;
      p0[v] = e0; p1[v] = e1;
    }

    // rescale running output (rows match C layout)
#pragma unroll
    for (int j = 0; j < 4; ++j)
#pragma unroll
      for (int v = 0; v < 8; ++v)
        acc[j][v] *= al[v];

    // ---- P: C-layout -> A-layout via LDS lane transpose (wave-private) ----
    half_t* pl = plds[wave];
#pragma unroll
    for (int v = 0; v < 8; ++v) {
      pl[(coff + v) * 32 + colL]      = (half_t)p0[v];
      pl[(coff + v) * 32 + colL + 16] = (half_t)p1[v];
    }
    asm volatile("s_wait_dscnt 0x0" ::: "memory");  // flush wave's DS stores
    v16h pa = loadA16x32(pl, 32, lane);

    // ---- acc += P @ V (V^T tile from LDS: pairs contiguous over keys) ----
#pragma unroll
    for (int j = 0; j < 4; ++j) {
      v16h bv = loadBT(lv + (j * 16) * 32, 32, lane);
      acc[j] = wmma_f16(pa, bv, acc[j]);
    }
  }

  // ---- epilogue: divide by row sums, store context f16 (b,seq,h*64+dk) ----
  int b = bh >> 4, h = bh & 15;
  half_t* cb = ctx + ((size_t)b * NSEQ + q0) * DM + h * DK;
#pragma unroll
  for (int v = 0; v < 8; ++v) {
    float inv = (rl[v] > 0.f) ? 1.0f / rl[v] : 0.0f;
#pragma unroll
    for (int j = 0; j < 4; ++j)
      cb[(size_t)(coff + v) * DM + j * 16 + colL] = (half_t)(acc[j][v] * inv);
  }
}

// ---------------- host launcher ----------------
extern "C" void kernel_launch(void* const* d_in, const int* in_sizes, int n_in,
                              void* d_out, int out_size, void* d_ws, size_t ws_size,
                              hipStream_t stream) {
  const float*         x    = (const float*)d_in[0];
  const unsigned char* mask = (const unsigned char*)d_in[1];  // jax bool -> 1 byte
  const float*         attw = (const float*)d_in[2];
  const float* Wq = (const float*)d_in[3];
  const float* bq = (const float*)d_in[4];
  const float* Wk = (const float*)d_in[5];
  const float* bk = (const float*)d_in[6];
  const float* Wv = (const float*)d_in[7];
  const float* bv = (const float*)d_in[8];
  const float* Wo = (const float*)d_in[9];
  const float* bo = (const float*)d_in[10];
  float* out = (float*)d_out;

  char* ws = (char*)d_ws;
  half_t* xh  = (half_t*)(ws + ((size_t)0  << 20));  // 8 MB  x f16
  half_t* wqh = (half_t*)(ws + ((size_t)8  << 20));  // 2 MB
  half_t* wkh = (half_t*)(ws + ((size_t)10 << 20));  // 2 MB
  half_t* wvh = (half_t*)(ws + ((size_t)12 << 20));  // 2 MB
  half_t* woh = (half_t*)(ws + ((size_t)14 << 20));  // 2 MB
  half_t* Qh  = (half_t*)(ws + ((size_t)16 << 20));  // 8 MB  (b,h,seq,dk)
  half_t* Kh  = (half_t*)(ws + ((size_t)24 << 20));  // 8 MB  (b,h,seq,dk)
  half_t* Vth = (half_t*)(ws + ((size_t)32 << 20));  // 8 MB  (b,h,dk,seq)
  half_t* ctx = (half_t*)(ws + ((size_t)40 << 20));  // 8 MB  (b,seq,dm)

  int M = BB * NSEQ;                      // 4096
  int nx = M * DM, nw = DM * DM;
  f32_to_f16<<<(nx + 255) / 256, 256, 0, stream>>>(x,  xh,  nx);
  f32_to_f16<<<(nw + 255) / 256, 256, 0, stream>>>(Wq, wqh, nw);
  f32_to_f16<<<(nw + 255) / 256, 256, 0, stream>>>(Wk, wkh, nw);
  f32_to_f16<<<(nw + 255) / 256, 256, 0, stream>>>(Wv, wvh, nw);
  f32_to_f16<<<(nw + 255) / 256, 256, 0, stream>>>(Wo, woh, nw);

  dim3 gg(DM / 64, M / (16 * GEMM_WAVES));
  gemm_xwt<<<gg, 32 * GEMM_WAVES, 0, stream>>>(xh, wqh, bq, Qh,  M, DM, DM, 0);
  gemm_xwt<<<gg, 32 * GEMM_WAVES, 0, stream>>>(xh, wkh, bk, Kh,  M, DM, DM, 0);
  gemm_xwt<<<gg, 32 * GEMM_WAVES, 0, stream>>>(xh, wvh, bv, Vth, M, DM, DM, 1);

  int nwaves = BB * HH * QT;              // 4096 q-tiles
  attn_kernel<<<nwaves / ATT_WAVES, 32 * ATT_WAVES, 0, stream>>>(
      Qh, Kh, Vth, attw, mask, ctx);

  gemm_xwt<<<gg, 32 * GEMM_WAVES, 0, stream>>>(ctx, woh, bo, out, M, DM, DM, 2);
}
